// DotProductAttention_54631984005167
// MI455X (gfx1250) — matrix operand order repaired and checked by reading, not compiled, not run on hardware
//
#include <hip/hip_runtime.h>

// ---- problem constants (from the reference) ----
#define BB 2
#define HH 16
#define SS 2048
#define DD 64
#define SW (SS / 32)   // mask words per row = 64

typedef __attribute__((ext_vector_type(16))) _Float16 v16h;
typedef __attribute__((ext_vector_type(8)))  _Float16 v8h;
typedef __attribute__((ext_vector_type(8)))  float    v8f;

#if __has_builtin(__builtin_amdgcn_exp2f)
#define EXP2F(x) __builtin_amdgcn_exp2f(x)
#else
#define EXP2F(x) exp2f(x)
#endif

__device__ __forceinline__ v8f wmma32(v16h a, v16h b, v8f c) {
  // D = A(16x32 f16) * B(32x16 f16) + C(16x16 f32)
  return __builtin_amdgcn_wmma_f32_16x16x32_f16(false, a, false, b, (short)0, c,
                                                false, false);
}

__device__ __forceinline__ v16h pack16(v8h lo, v8h hi) {
  v16h r;
#pragma unroll
  for (int i = 0; i < 8; ++i) { r[i] = lo[i]; r[i + 8] = hi[i]; }
  return r;
}

// K-permutation baked into Vt columns: column Cv(k) holds key k within each
// 32-key block, matching the P A-operand's packed-pair storage order.
__device__ __forceinline__ int cv_perm(int k) {
  return ((k & 4) << 1) | ((k & 8) << 1) | ((k & 2) << 1) | ((k & 1) << 1) |
         ((k & 16) >> 4);
}

// ---- pre-pass 1: pack int mask -> 1 bit/element, one word per wave (wave32) ----
__launch_bounds__(256)
__global__ void pack_mask_gfx1250(const int* __restrict__ M,
                                  unsigned int* __restrict__ W) {
  const int gw   = (blockIdx.x * blockDim.x + threadIdx.x) >> 5; // global wave id
  const int lane = threadIdx.x & 31;
  const int m    = M[(size_t)gw * 32 + lane];
  const unsigned long long bal = __ballot(m != 0);   // wave32: low 32 bits valid
  if (lane == 0) W[gw] = (unsigned int)bal;
}

// ---- pre-pass 2: K f32 -> f16 row-major (one-time convert) ----
__launch_bounds__(256)
__global__ void cvt_k_gfx1250(const float* __restrict__ K,
                              _Float16* __restrict__ Kh) {
  const size_t g = ((size_t)blockIdx.x * blockDim.x + threadIdx.x) * 8;
  float4 a = *(const float4*)(K + g);
  float4 b = *(const float4*)(K + g + 4);
  _Float16 t[8] = { (_Float16)a.x, (_Float16)a.y, (_Float16)a.z, (_Float16)a.w,
                    (_Float16)b.x, (_Float16)b.y, (_Float16)b.z, (_Float16)b.w };
  *(v8h*)(Kh + g) = *(const v8h*)t;
}

// ---- pre-pass 3: V f32 -> f16 TRANSPOSED + K-permuted:
//      Vt[bh][d][(s & ~31) | Cv(s & 31)] = V[bh][s][d] ----
__launch_bounds__(256)
__global__ void cvt_vt_gfx1250(const float* __restrict__ V,
                               _Float16* __restrict__ Vt) {
  const size_t g  = (size_t)blockIdx.x * blockDim.x + threadIdx.x; // one per 8 elems
  const int    bh = (int)(g / (SS * 8));
  const int    rm = (int)(g % (SS * 8));
  const int    s  = rm >> 3;
  const int    dc = (rm & 7) * 8;
  const int    sp = (s & ~31) | cv_perm(s & 31);     // permuted column
  const float* vp = V + ((size_t)bh * SS + s) * DD + dc;
  float4 a = *(const float4*)vp;
  float4 b = *(const float4*)(vp + 4);
  float vv[8] = { a.x, a.y, a.z, a.w, b.x, b.y, b.z, b.w };
  _Float16* o = Vt + (size_t)bh * DD * SS + sp;      // + d*SS
#pragma unroll
  for (int i = 0; i < 8; ++i) o[(size_t)(dc + i) * SS] = (_Float16)vv[i];
}

__launch_bounds__(128)
__global__ void fa_fwd_gfx1250(const float* __restrict__ Q,
                               const _Float16* __restrict__ Kh,
                               const _Float16* __restrict__ Vt,
                               const unsigned int* __restrict__ MW,
                               float* __restrict__ O) {
  constexpr int   BM     = 128;    // query rows per block (4 waves x 32)
  constexpr int   BN     = 32;     // key columns per iteration
  // fold 1/sqrt(64) AND log2(e) into Q so P = exp2(S) directly (v_exp_f32)
  constexpr float QSCALE = 0.125f * 1.44269504088896340736f;

  // K block, row-major f16, padded row stride 72 (144B rows, 16B-aligned chunks)
  __shared__ __align__(16) _Float16 Klds[32 * 72];
  // V block TRANSPOSED (key columns pre-permuted): VldsT[d][q], stride 40
  __shared__ __align__(16) _Float16 VldsT[64 * 40];
  // per-wave P tiles (32 rows each); row holds packed (k, k+16) pairs
  __shared__ __align__(16) _Float16 Plds[4 * 32 * 40];

  const int tid  = threadIdx.x;
  const int w    = tid >> 5;    // wave id 0..3
  const int lane = tid & 31;
  const int hl   = lane >> 4;   // half-wave: 0 or 1
  const int lr   = lane & 15;

  const int bh      = blockIdx.y;       // b*H + h
  const int b       = bh / HH;
  const int qblk    = blockIdx.x * BM;
  const int rowbase = qblk + w * 32;    // first query row of this wave (32 rows)

  const float*        Qb  = Q  + (size_t)bh * SS * DD;
  const _Float16*     Khb = Kh + (size_t)bh * SS * DD;
  const _Float16*     Vtb = Vt + (size_t)bh * DD * SS;
  const unsigned int* Mb  = MW + (size_t)b  * SS * SW;
  float*              Ob  = O  + (size_t)bh * SS * DD;

  // ---- load this wave's 32x64 Q stripe (pre-scaled) into A-layout tiles ----
  // A layout (16-bit, 16x32): lane -> M = lr; element i -> K = (i/8)*16 + hl*8 + i%8
  v16h qlo[2], qhi[2];
#pragma unroll
  for (int g = 0; g < 2; ++g) {
    const float* qr = Qb + (size_t)(rowbase + 16 * g + lr) * DD;
    _Float16 t[32];
#pragma unroll
    for (int c = 0; c < 4; ++c) {
      const float* p = qr + c * 16 + hl * 8;
      float4 a = *(const float4*)p;
      float4 bq = *(const float4*)(p + 4);
      t[c*8+0] = (_Float16)(a.x * QSCALE);  t[c*8+1] = (_Float16)(a.y * QSCALE);
      t[c*8+2] = (_Float16)(a.z * QSCALE);  t[c*8+3] = (_Float16)(a.w * QSCALE);
      t[c*8+4] = (_Float16)(bq.x * QSCALE); t[c*8+5] = (_Float16)(bq.y * QSCALE);
      t[c*8+6] = (_Float16)(bq.z * QSCALE); t[c*8+7] = (_Float16)(bq.w * QSCALE);
    }
#pragma unroll
    for (int i = 0; i < 16; ++i) { qlo[g][i] = t[i]; qhi[g][i] = t[16 + i]; }
  }

  // constant ones B-matrix for row-sum WMMA (permutation-invariant)
  v16h ones;
#pragma unroll
  for (int i = 0; i < 16; ++i) ones[i] = (_Float16)1.0f;

  // accumulators: acc[g][t] covers d = 16t + lr; l_acc[g][r] = running row sum
  v8f acc[2][4] = {};
  v8f l_acc[2]  = {};

  // staging assignments (pure f16 block copies)
  const int krow = tid >> 2;        // K: row in block, 0..31
  const int kcol = (tid & 3) * 16;  // K: first of 16 d-columns
  const int vrow = tid >> 1;        // Vt: d row, 0..63
  const int vcol = (tid & 1) * 16;  // Vt: first of 16 key-columns

  for (int kb = 0; kb < SS; kb += BN) {
    // ---- cooperative stage: Kh block and Vt block -> LDS (f16 copies) ----
    {
      const _Float16* kg = Khb + (size_t)(kb + krow) * DD + kcol;
      _Float16* kd = &Klds[krow * 72 + kcol];
      *(v8h*)kd       = *(const v8h*)kg;
      *(v8h*)(kd + 8) = *(const v8h*)(kg + 8);
      const _Float16* vg = Vtb + (size_t)vrow * SS + kb + vcol;
      _Float16* vd = &VldsT[vrow * 40 + vcol];
      *(v8h*)vd       = *(const v8h*)vg;
      *(v8h*)(vd + 8) = *(const v8h*)(vg + 8);
      if (kb + BN < SS) {
        __builtin_prefetch(kg + (size_t)BN * DD, 0, 3);  // global_prefetch_b8
        __builtin_prefetch(vg + BN, 0, 3);
      }
    }
    __syncthreads();

    // ---- load K B-tiles once, reuse across both row groups ----
    // B layout (16-bit, 32x16): lane -> N = lr; element i -> K = (i/8)*16 + hl*8 + i%8
    const _Float16* k0r = &Klds[(lr)      * 72 + hl * 8];
    const _Float16* k1r = &Klds[(16 + lr) * 72 + hl * 8];
    const v16h b00 = pack16(*(const v8h*)(k0r),      *(const v8h*)(k0r + 16));
    const v16h b01 = pack16(*(const v8h*)(k0r + 32), *(const v8h*)(k0r + 48));
    const v16h b10 = pack16(*(const v8h*)(k1r),      *(const v8h*)(k1r + 16));
    const v16h b11 = pack16(*(const v8h*)(k1r + 32), *(const v8h*)(k1r + 48));

#pragma unroll
    for (int g = 0; g < 2; ++g) {
      // ---- S = (Q*log2e/8) * K^T : two 16x16 f32 tiles ----
      v8f s0 = {}, s1 = {};
      s0 = wmma32(qlo[g], b00, s0);
      s0 = wmma32(qhi[g], b01, s0);
      s1 = wmma32(qlo[g], b10, s1);
      s1 = wmma32(qhi[g], b11, s1);

      // ---- P = mask ? exp2(S) : 0, packed pair + bitwise mask, one b32 store ----
#pragma unroll
      for (int r = 0; r < 8; ++r) {
        const int qrow = rowbase + 16 * g + 8 * hl + r;
        const unsigned int mw  = Mb[(size_t)qrow * SW + (kb >> 5)];
        const unsigned int m32 = ((mw >> lr) & 0x10001u) * 0xFFFFu;
        auto pk = __builtin_amdgcn_cvt_pkrtz(EXP2F(s0[r]), EXP2F(s1[r]));
        unsigned int u;
        __builtin_memcpy(&u, &pk, 4);
        u &= m32;   // zero masked f16 halves (AND 0 -> +0.0)
        *(unsigned int*)&Plds[(w * 32 + 16 * g + 8 * hl + r) * 40 + 2 * lr] = u;
      }
    }
    // same-wave LDS RAW across lanes: DS ops are in-order per wave; wait + fence
    asm volatile("s_wait_dscnt 0" ::: "memory");

    // ---- O += P * V ; l += P * ones  (row sums on the matrix unit) ----
    {
      v16h pa[2];
#pragma unroll
      for (int g = 0; g < 2; ++g) {
        const _Float16* par = &Plds[(w * 32 + 16 * g + lr) * 40 + hl * 16];
        pa[g] = pack16(*(const v8h*)par, *(const v8h*)(par + 8)); // packed-pair order
      }
      l_acc[0] = wmma32(pa[0], ones, l_acc[0]);
      l_acc[1] = wmma32(pa[1], ones, l_acc[1]);
#pragma unroll
      for (int t = 0; t < 4; ++t) {
        const _Float16* vr = &VldsT[(t * 16 + lr) * 40 + hl * 16];
        const v16h vb = pack16(*(const v8h*)(vr), *(const v8h*)(vr + 8)); // permuted
        acc[0][t] = wmma32(pa[0], vb, acc[0][t]);
        acc[1][t] = wmma32(pa[1], vb, acc[1][t]);
      }
    }
    __syncthreads();   // protect Klds/VldsT before next stage
  }

  // ---- epilogue: O = acc / l ----
#pragma unroll
  for (int g = 0; g < 2; ++g)
#pragma unroll
    for (int r = 0; r < 8; ++r) {
      const float inv = 1.0f / l_acc[g][r];
      float* orow = Ob + (size_t)(rowbase + 16 * g + 8 * hl + r) * DD;
#pragma unroll
      for (int t = 0; t < 4; ++t) orow[t * 16 + lr] = acc[g][t][r] * inv;
    }
}

extern "C" void kernel_launch(void* const* d_in, const int* in_sizes, int n_in,
                              void* d_out, int out_size, void* d_ws, size_t ws_size,
                              hipStream_t stream) {
  (void)in_sizes; (void)n_in; (void)out_size; (void)ws_size;
  const float* q    = (const float*)d_in[0];
  const float* k    = (const float*)d_in[1];
  const float* v    = (const float*)d_in[2];
  const int*   mask = (const int*)d_in[3];
  float*       out  = (float*)d_out;

  // d_ws layout: [ mask bits 1MB | Kh f16 8.39MB | Vt f16 8.39MB ]
  char* ws = (char*)d_ws;
  unsigned int* mw = (unsigned int*)ws;                       // 1,048,576 B
  _Float16*     kh = (_Float16*)(ws + (1u << 20));            // 8,388,608 B
  _Float16*     vt = (_Float16*)(ws + (1u << 20) + (BB*HH*SS*DD*2));

  const int nelem  = BB * HH * SS * DD;     // 4,194,304
  const int nwords = BB * SS * SW;          // 262,144

  pack_mask_gfx1250<<<dim3(nwords / 8), dim3(256), 0, stream>>>(mask, mw);
  cvt_k_gfx1250<<<dim3(nelem / (256 * 8)), dim3(256), 0, stream>>>(k, kh);
  cvt_vt_gfx1250<<<dim3(nelem / (256 * 8)), dim3(256), 0, stream>>>(v, vt);

  dim3 grid(SS / 128, BB * HH);   // 16 x 32 blocks
  dim3 block(128);                // 4 wave32 waves, 32 query rows each
  fa_fwd_gfx1250<<<grid, block, 0, stream>>>(q, kh, vt, mw, out);
}
